// AffConv_63058709840232
// MI455X (gfx1250) — compile-verified, matrix operand order checked by reading, not run on hardware
//
#include <hip/hip_runtime.h>

// ---------------- CDNA5 WMMA types ----------------
typedef __attribute__((ext_vector_type(16))) __bf16 v16bf;
typedef __attribute__((ext_vector_type(8)))  float  v8f;

#define NGRAPH 4
#define MPTS   50000
#define INC    64
#define KNN    9
#define OUTC   64
#define NKT    21                 // 672 / 32 k-tiles (658 valid, padded)
#define TILES_PER_G (MPTS/16)     // 3125
#define WAVES_PER_BLOCK 20        // 640 threads; 12500 tiles / 20 = 625 blocks exact
#define THREADS_PER_BLOCK (WAVES_PER_BLOCK*32)
#define NFRAGS (NKT*4*2*32)       // 5376 v16bf fragments
#define WPACK_BYTES (NFRAGS*32)   // 172032 B (fits 320 KB WGP LDS)
#define DISTP1 11.0f              // dist + 1.0

// RNE bf16 (used only in the one-shot W pack kernel)
__device__ __forceinline__ unsigned short bf16_rne(float x) {
    unsigned u = __float_as_uint(x);
    u += 0x7FFFu + ((u >> 16) & 1u);
    return (unsigned short)(u >> 16);
}
__device__ __forceinline__ float bf16_to_f32(unsigned short h) {
    return __uint_as_float(((unsigned)h) << 16);
}

union AFrag { v16bf v; unsigned u[8]; unsigned short h[16]; };

// Permuted-K -> original W row. Reference x layout per row:
//   for nb in 0..8: [2 locs][64 feats]  (rows nb*66+0..65), then center 594..657
// Our permuted K:  [0,576) neighbor feats | [576,640) center | [640,658) locs | pad
__device__ __forceinline__ int w_orig_row(int k) {
    if (k < 576) { int nb = k >> 6, c = k & 63; return nb * 66 + 2 + c; }
    if (k < 640) { return 594 + (k - 576); }
    int t = k - 640; return (t >> 1) * 66 + (t & 1);
}

// ---------------- W pre-pack: fragment-order bf16 hi/lo (RNE) ----------------
// Layout: wp[((kt*4 + nt)*2 + sel)*32 + lane] = v16bf fragment (32B each)
__global__ void affconv_pack_w(const float* __restrict__ W, v16bf* __restrict__ wp) {
    int tid = blockIdx.x * blockDim.x + threadIdx.x;
    if (tid >= NKT * 4 * 32) return;
    int lane = tid & 31;
    int nt   = (tid >> 5) & 3;
    int kt   = tid >> 7;
    int n    = nt * 16 + (lane & 15);

    AFrag hi, lo;
#pragma unroll
    for (int j = 0; j < 16; j++) {
        int kl = j + 8 * ((j >> 3) + (lane >> 4));   // B mirrors A's 16-bit K layout
        int k  = kt * 32 + kl;
        float w = 0.0f;
        if (k < 658) w = W[w_orig_row(k) * OUTC + n];
        unsigned short h = bf16_rne(w);
        hi.h[j] = h;
        lo.h[j] = bf16_rne(w - bf16_to_f32(h));
    }
    size_t base = (size_t)((kt * 4 + nt) * 2) * 32 + lane;
    wp[base]      = hi.v;
    wp[base + 32] = lo.v;
}

// ---------------- cheap truncating bf16x2 split for A (hot loop) ----------------
// hi = trunc16(x); lo = trunc16(x - f32(hi)).  Residual ~2^-16 relative, same
// order as the dropped alo*blo term.  Packing via v_perm_b32: 1 op / dword.
__device__ __forceinline__ void split16(const float fv[16], AFrag& ahi, AFrag& alo) {
    unsigned ulo[16];
#pragma unroll
    for (int j = 0; j < 16; j++) {
        unsigned u = __float_as_uint(fv[j]);
        float lo = fv[j] - __uint_as_float(u & 0xFFFF0000u);   // vectorizes to v_pk_add_f32
        ulo[j] = __float_as_uint(lo);
    }
#pragma unroll
    for (int d = 0; d < 8; d++) {
        unsigned u0 = __float_as_uint(fv[2 * d]);
        unsigned u1 = __float_as_uint(fv[2 * d + 1]);
        ahi.u[d] = __builtin_amdgcn_perm(u1, u0, 0x07060302u);        // {u1.hi16, u0.hi16}
        alo.u[d] = __builtin_amdgcn_perm(ulo[2 * d + 1], ulo[2 * d], 0x07060302u);
    }
}

__device__ __forceinline__ void mm3(const AFrag& ahi, const AFrag& alo,
                                    const v16bf* wlds, int kt, int lane,
                                    v8f acc[4]) {
#pragma unroll
    for (int nt = 0; nt < 4; nt++) {
        int fb = ((kt * 4 + nt) * 2) * 32 + lane;
        v16bf bh = wlds[fb];        // ds_load_b128 x2
        v16bf bl = wlds[fb + 32];
        v8f a = acc[nt];
        a = __builtin_amdgcn_wmma_f32_16x16x32_bf16(false, ahi.v, false, bh, (short)0, a, false, false);
        a = __builtin_amdgcn_wmma_f32_16x16x32_bf16(false, alo.v, false, bh, (short)0, a, false, false);
        a = __builtin_amdgcn_wmma_f32_16x16x32_bf16(false, ahi.v, false, bl, (short)0, a, false, false);
        acc[nt] = a;
    }
}

// load one 32-wide half-row worth of A elements for this lane, split, accumulate
__device__ __forceinline__ void do_feat_tile(const float* base, const v16bf* wlds,
                                             int kt, int lane, v8f acc[4]) {
    float4 f0 = *(const float4*)(base + 0);
    float4 f1 = *(const float4*)(base + 4);
    float4 f2 = *(const float4*)(base + 16);
    float4 f3 = *(const float4*)(base + 20);
    float fv[16] = {f0.x, f0.y, f0.z, f0.w, f1.x, f1.y, f1.z, f1.w,
                    f2.x, f2.y, f2.z, f2.w, f3.x, f3.y, f3.z, f3.w};
    AFrag ahi, alo;
    split16(fv, ahi, alo);
    mm3(ahi, alo, wlds, kt, lane, acc);
}

// ---------------- fused gather + GEMM + bias + ReLU ----------------
__global__ __launch_bounds__(THREADS_PER_BLOCK) void affconv_gemm(
    const float* __restrict__ feats, const int* __restrict__ aff,
    const float* __restrict__ locs,  const v16bf* __restrict__ wp,
    const float* __restrict__ bias,  float* __restrict__ out)
{
    extern __shared__ __align__(16) char smem_raw[];
    v16bf* wlds = (v16bf*)smem_raw;

    // Stage the entire packed W (172 KB) into LDS once per workgroup.
    {
        const float4* src = (const float4*)wp;
        float4*       dst = (float4*)smem_raw;
        for (int i = threadIdx.x; i < WPACK_BYTES / 16; i += THREADS_PER_BLOCK)
            dst[i] = src[i];
    }
    __syncthreads();

    const int lane = threadIdx.x & 31;
    const int wave = threadIdx.x >> 5;
    const int tile = blockIdx.x * WAVES_PER_BLOCK + wave;  // 0..12499, exact
    const int g    = tile / TILES_PER_G;
    const int trow = (tile % TILES_PER_G) * 16;
    const int m    = trow + (lane & 15);                   // A-matrix row for this lane
    const size_t gbase = (size_t)g * MPTS;

    // per-row gather indices + center loc (lanes L and L+16 duplicate -> fine)
    int nidx[KNN];
#pragma unroll
    for (int nb = 0; nb < KNN; nb++) nidx[nb] = aff[(gbase + m) * KNN + nb];
    const float cx = locs[(gbase + m) * 2 + 0];
    const float cy = locs[(gbase + m) * 2 + 1];

    v8f acc[4] = {v8f{}, v8f{}, v8f{}, v8f{}};
    const int kofs = (lane >= 16) ? 8 : 0;                 // K sub-offset per A layout

    // neighbor k-tiles 0..17: both 32-wide halves of each gathered row together
    // (one clause fetches the lane's share of the full 256 B row)
    for (int nb = 0; nb < KNN; nb++) {
        const float* rowp = feats + (gbase + (size_t)nidx[nb]) * INC + kofs;
        do_feat_tile(rowp,      wlds, nb * 2,     lane, acc);
        do_feat_tile(rowp + 32, wlds, nb * 2 + 1, lane, acc);
    }

    // center k-tiles 18,19
    {
        const float* rowp = feats + (gbase + (size_t)m) * INC + kofs;
        do_feat_tile(rowp,      wlds, 18, lane, acc);
        do_feat_tile(rowp + 32, wlds, 19, lane, acc);
    }

    // k-tile 20: clipped relative locs (k 640..657 valid, rest zero pad)
    {
        float fv[16];
#pragma unroll
        for (int j = 0; j < 16; j++) {
            int kl = j + 8 * ((j >> 3) + (lane >> 4));
            int k  = 640 + kl;
            float v = 0.0f;
            if (k < 658) {
                int t = k - 640, nb = t >> 1, xy = t & 1;
                float nl = locs[(gbase + (size_t)nidx[nb]) * 2 + xy];
                float cl = xy ? cy : cx;
                v = (nl - cl) * (1.0f / DISTP1);
                v = fminf(fmaxf(v, -1.0f), 1.0f);
            }
            fv[j] = v;
        }
        AFrag ahi, alo;
        split16(fv, ahi, alo);
        mm3(ahi, alo, wlds, 20, lane, acc);
    }

    // epilogue: C/D layout -> lane covers col n = nt*16 + (lane&15),
    // rows m = trow + 8*(lane>=16) + r ; lanes 0..15 store 64B-contiguous runs
    const int rbase = trow + 8 * (lane >> 4);
#pragma unroll
    for (int nt = 0; nt < 4; nt++) {
        int n = nt * 16 + (lane & 15);
        float bb = bias[n];
#pragma unroll
        for (int r = 0; r < 8; r++) {
            float v = acc[nt][r] + bb;
            out[(gbase + rbase + r) * OUTC + n] = fmaxf(v, 0.0f);
        }
    }
}

extern "C" void kernel_launch(void* const* d_in, const int* in_sizes, int n_in,
                              void* d_out, int out_size, void* d_ws, size_t ws_size,
                              hipStream_t stream) {
    const float* feats = (const float*)d_in[0];   // (4, 50000, 64) f32
    const int*   aff   = (const int*)  d_in[1];   // (4, 50000, 9)  i32
    const float* locs  = (const float*)d_in[2];   // (4, 50000, 2)  f32
    const float* W     = (const float*)d_in[3];   // (658, 64)      f32
    const float* b     = (const float*)d_in[4];   // (64,)          f32
    float* out = (float*)d_out;                   // (4, 50000, 64) f32
    v16bf* wp  = (v16bf*)d_ws;                    // 172032 B scratch

    // 1) pack W into per-lane bf16 hi/lo fragments (runs every call; tiny)
    int packThreads = NKT * 4 * 32;               // 2688
    affconv_pack_w<<<(packThreads + 127) / 128, 128, 0, stream>>>(W, wp);

    // 2) fused gather + bf16x2 WMMA GEMM + bias + ReLU, W staged in LDS
    int nBlocks = (NGRAPH * TILES_PER_G) / WAVES_PER_BLOCK;   // 625
    affconv_gemm<<<nBlocks, THREADS_PER_BLOCK, WPACK_BYTES, stream>>>(
        feats, aff, locs, wp, b, out);
}
